// SimpleTemporalGNN_43267500540704
// MI455X (gfx1250) — compile-verified, matrix operand order checked by reading, not run on hardware
//
#include <hip/hip_runtime.h>
#include <hip/hip_bf16.h>

// ---------------------------------------------------------------------------
// CDNA5 (gfx1250) WMMA types
// ---------------------------------------------------------------------------
typedef __attribute__((ext_vector_type(16))) __bf16 v16bf;
typedef __attribute__((ext_vector_type(8)))  float  v8f;

union Frag {
    unsigned u[8];
    uint4    q[2];
    v16bf    v;
};

__device__ __forceinline__ unsigned bf16_bits(float f) {
    unsigned a = __builtin_bit_cast(unsigned, f);
    return (a + 0x7FFFu + ((a >> 16) & 1u)) >> 16;   // RNE truncate to bf16
}

__device__ __forceinline__ unsigned pack_bf16x2(float lo, float hi) {
    return bf16_bits(lo) | (bf16_bits(hi) << 16);
}

// ---------------------------------------------------------------------------
// Weight pre-pack: fragment-order bf16 pairs.
//   idx = ((ks*ntiles + nt)*32 + lane)*8 + v
//   lane<16: K {0..7} in v0..3, {16..23} in v4..7 ; lane>=16: +8 / +24
//   B stored "column-major mirror": lane holds column n = nt*16 + (lane&15)
// ---------------------------------------------------------------------------
__global__ __launch_bounds__(256)
void pack_weights_kernel(const float* __restrict__ W, unsigned* __restrict__ out,
                         int ncols, int transposed, int ldW) {
    const int ntiles = ncols >> 4;
    const int total  = 4 * ntiles * 32 * 8;   // ksteps=4 (K=128)
    for (int idx = blockIdx.x * 256 + threadIdx.x; idx < total;
         idx += gridDim.x * 256) {
        const int v    = idx & 7;
        const int lane = (idx >> 3) & 31;
        const int nt   = (idx >> 8) % ntiles;
        const int ks   = idx / (ntiles << 8);
        const int half = lane >> 4;
        const int n    = nt * 16 + (lane & 15);
        const int k    = ks * 32 + ((v & 4) << 2) + half * 8 + ((v & 3) << 1);
        float f0, f1;
        if (transposed) { f0 = W[n * ldW + k];     f1 = W[n * ldW + k + 1]; }
        else            { f0 = W[k * ldW + n];     f1 = W[(k + 1) * ldW + n]; }
        out[idx] = pack_bf16x2(f0, f1);
    }
}

// ---------------------------------------------------------------------------
// WMMA inner pass: one wave, 16 rows x (8 tiles = 128 cols), K=128.
// A row-major bf16; software-pipelined: next fragment loads issued before
// the current WMMA so waits only cover older loads.
// ---------------------------------------------------------------------------
__device__ __forceinline__
void wmma_pass_k128(const unsigned short* __restrict__ Arow,
                    const unsigned* __restrict__ Bp,
                    int ntiles, int colgrp, int half, int lane, v8f acc[8]) {
    auto loadB = [&](int ks, int nt) {
        Frag b;
        const unsigned* bp =
            Bp + (((size_t)(ks * ntiles + colgrp * 8 + nt) * 32 + lane) * 8);
        b.q[0] = *(const uint4*)(bp);
        b.q[1] = *(const uint4*)(bp + 4);
        return b;
    };
    auto loadA = [&](int ks) {
        Frag a;
        const unsigned short* ap = Arow + ks * 32 + half * 8;
        a.q[0] = *(const uint4*)(ap);        // K base+0 .. +7
        a.q[1] = *(const uint4*)(ap + 16);   // K base+16 .. +23
        return a;
    };

    Frag a_cur = loadA(0);
    Frag b_cur = loadB(0, 0);
#pragma unroll
    for (int it = 0; it < 32; ++it) {
        const int nt = it & 7;
        Frag a_nxt = a_cur;
        Frag b_nxt = b_cur;
        if (it < 31) {
            const int ks2 = (it + 1) >> 3;
            const int nt2 = (it + 1) & 7;
            b_nxt = loadB(ks2, nt2);
            if (nt2 == 0) a_nxt = loadA(ks2);
        }
        acc[nt] = __builtin_amdgcn_wmma_f32_16x16x32_bf16(
            false, a_cur.v, false, b_cur.v, (short)0, acc[nt], false, false);
        a_cur = a_nxt;
        b_cur = b_nxt;
    }
}

// ---------------------------------------------------------------------------
// GCN GEMM: C[N,128] = Abf16[N,128] @ Bpacked(128x128). wave = 16-row strip.
// ---------------------------------------------------------------------------
__global__ __launch_bounds__(256)
void gemm_n128_kernel(const unsigned short* __restrict__ A,
                      const unsigned* __restrict__ Bp,
                      float* __restrict__ C, int nRows) {
    const int wave  = threadIdx.x >> 5;
    const int lane  = threadIdx.x & 31;
    const int strip = blockIdx.x * 8 + wave;
    const int row0  = strip * 16;
    if (row0 >= nRows) return;
    const int half = lane >> 4;
    const int rloc = lane & 15;
    int arow = row0 + rloc;
    if (arow >= nRows) arow = nRows - 1;

    const v8f vzero = {0.f, 0.f, 0.f, 0.f, 0.f, 0.f, 0.f, 0.f};
    v8f acc[8];
#pragma unroll
    for (int i = 0; i < 8; ++i) acc[i] = vzero;

    wmma_pass_k128(A + (size_t)arow * 128, Bp, 8, 0, half, lane, acc);

#pragma unroll
    for (int nt = 0; nt < 8; ++nt) {
        const int col = nt * 16 + rloc;
#pragma unroll
        for (int v = 0; v < 8; ++v) {
            const int r = row0 + half * 8 + v;
            if (r < nRows) C[(size_t)r * 128 + col] = acc[nt][v];
        }
    }
}

// ---------------------------------------------------------------------------
// Fused LSTM step: gates = emb@W_ih^T + h@W_hh^T (WMMA, bf16 A) -> LDS ->
// cell update. Block = 32 rows; wave w: rowhalf = w&1, colgroup = w>>1.
// Each block touches only its own rows of h/c/h_bf -> in-place safe.
// ---------------------------------------------------------------------------
__global__ __launch_bounds__(256)
void lstm_step_kernel(const unsigned short* __restrict__ emb_bf,
                      const unsigned short* __restrict__ h_bf_in,
                      const unsigned* __restrict__ Wihp,
                      const unsigned* __restrict__ Whhp,
                      const float* __restrict__ b_ih,
                      const float* __restrict__ b_hh,
                      float* __restrict__ h, float* __restrict__ c,
                      unsigned short* __restrict__ h_bf_out, int nRows) {
    __shared__ float gbuf[32 * 512];
    const int tid     = threadIdx.x;
    const int wave    = tid >> 5;
    const int lane    = tid & 31;
    const int rowhalf = wave & 1;
    const int colgrp  = wave >> 1;
    const int rowBase = blockIdx.x * 32;
    const int row0    = rowBase + rowhalf * 16;
    const int half    = lane >> 4;
    const int rloc    = lane & 15;
    int arow = row0 + rloc;
    if (arow >= nRows) arow = nRows - 1;

    const v8f vzero = {0.f, 0.f, 0.f, 0.f, 0.f, 0.f, 0.f, 0.f};
    v8f acc[8];
#pragma unroll
    for (int i = 0; i < 8; ++i) acc[i] = vzero;

    wmma_pass_k128(emb_bf  + (size_t)arow * 128, Wihp, 32, colgrp, half, lane, acc);
    wmma_pass_k128(h_bf_in + (size_t)arow * 128, Whhp, 32, colgrp, half, lane, acc);

#pragma unroll
    for (int nt = 0; nt < 8; ++nt) {
        const int cidx = colgrp * 128 + nt * 16 + rloc;
#pragma unroll
        for (int v = 0; v < 8; ++v) {
            const int r = rowhalf * 16 + half * 8 + v;
            gbuf[r * 512 + cidx] = acc[nt][v];
        }
    }
    __syncthreads();

    for (int i = tid; i < 32 * 128; i += 256) {
        const int r = i >> 7;
        const int j = i & 127;
        const int n = rowBase + r;
        if (n < nRows) {
            const float gi = gbuf[r * 512 + j]       + b_ih[j]       + b_hh[j];
            const float gf = gbuf[r * 512 + 128 + j] + b_ih[128 + j] + b_hh[128 + j];
            const float gg = gbuf[r * 512 + 256 + j] + b_ih[256 + j] + b_hh[256 + j];
            const float go = gbuf[r * 512 + 384 + j] + b_ih[384 + j] + b_hh[384 + j];
            const float si = 1.f / (1.f + __expf(-gi));
            const float sf = 1.f / (1.f + __expf(-gf));
            const float so = 1.f / (1.f + __expf(-go));
            const float tg = tanhf(gg);
            const size_t off = (size_t)n * 128 + j;
            const float cn = sf * c[off] + si * tg;
            const float hn = so * tanhf(cn);
            c[off] = cn;
            h[off] = hn;
            h_bf_out[off] = (unsigned short)bf16_bits(hn);
        }
    }
}

// ---------------------------------------------------------------------------
// Elementwise / edge kernels
// ---------------------------------------------------------------------------
__global__ void fill_f32_kernel(float* __restrict__ p, float val, long long n) {
    for (long long i = (long long)blockIdx.x * blockDim.x + threadIdx.x; i < n;
         i += (long long)gridDim.x * blockDim.x)
        p[i] = val;
}

// f32 -> bf16 (pair-packed stores)
__global__ void cvt_f32_bf16_kernel(const float* __restrict__ in,
                                    unsigned* __restrict__ out, long long npairs) {
    for (long long i = (long long)blockIdx.x * blockDim.x + threadIdx.x; i < npairs;
         i += (long long)gridDim.x * blockDim.x) {
        const float2 f = ((const float2*)in)[i];
        out[i] = pack_bf16x2(f.x, f.y);
    }
}

__global__ void deg_count_kernel(float* __restrict__ deg,
                                 const int* __restrict__ dst, int E) {
    const int e = blockIdx.x * blockDim.x + threadIdx.x;
    if (e < E) unsafeAtomicAdd(&deg[dst[e]], 1.0f);
}

__global__ void finish_dinv_kernel(float* __restrict__ dinv, int n) {
    const int i = blockIdx.x * blockDim.x + threadIdx.x;
    if (i < n) dinv[i] = rsqrtf(dinv[i]);
}

// wave-per-edge scatter: agg[dst] += h[src] * dinv[src]*dinv[dst]
__global__ __launch_bounds__(256)
void edge_scatter_kernel(const float* __restrict__ hsrc, float* __restrict__ agg,
                         const int* __restrict__ src, const int* __restrict__ dst,
                         const float* __restrict__ dinv, int E) {
    const int wave = threadIdx.x >> 5;
    const int lane = threadIdx.x & 31;
    const int e    = blockIdx.x * 8 + wave;
    if (e >= E) return;
    const int s = src[e];
    const int d = dst[e];
    const float norm = dinv[s] * dinv[d];
    const float4 hv = *(const float4*)(hsrc + (size_t)s * 128 + lane * 4);
    float* ap = agg + (size_t)d * 128 + lane * 4;
    unsafeAtomicAdd(ap + 0, hv.x * norm);
    unsafeAtomicAdd(ap + 1, hv.y * norm);
    unsafeAtomicAdd(ap + 2, hv.z * norm);
    unsafeAtomicAdd(ap + 3, hv.w * norm);
}

// out_bf16 = relu(agg + xw*dinv^2 + bias); two elements per thread, packed store
__global__ void gcn_combine_kernel(const float* __restrict__ agg,
                                   const float* __restrict__ xw,
                                   const float* __restrict__ dinv,
                                   const float* __restrict__ bias,
                                   unsigned* __restrict__ out_bf, int nRows) {
    const long long i = (long long)blockIdx.x * blockDim.x + threadIdx.x;
    if (i < (long long)nRows * 64) {              // pairs
        const long long e0 = i * 2;
        const int n = (int)(e0 >> 7);
        const int j = (int)(e0 & 127);
        const float di2 = dinv[n] * dinv[n];
        const float2 ag = ((const float2*)agg)[i];
        const float2 xv = ((const float2*)xw)[i];
        const float v0 = fmaxf(ag.x + xv.x * di2 + bias[j],     0.0f);
        const float v1 = fmaxf(ag.y + xv.y * di2 + bias[j + 1], 0.0f);
        out_bf[i] = pack_bf16x2(v0, v1);
    }
}

__global__ void copy_f32_kernel(const float* __restrict__ in,
                                float* __restrict__ out, long long n) {
    for (long long i = (long long)blockIdx.x * blockDim.x + threadIdx.x; i < n;
         i += (long long)gridDim.x * blockDim.x)
        out[i] = in[i];
}

// ---------------------------------------------------------------------------
// Launcher
// ---------------------------------------------------------------------------
extern "C" void kernel_launch(void* const* d_in, const int* in_sizes, int n_in,
                              void* d_out, int out_size, void* d_ws, size_t ws_size,
                              hipStream_t stream) {
    const float* node_features = (const float*)d_in[0];
    const int*   edge_index    = (const int*)d_in[1];
    const float* W1   = (const float*)d_in[2];
    const float* b1   = (const float*)d_in[3];
    const float* W2   = (const float*)d_in[4];
    const float* b2   = (const float*)d_in[5];
    const float* W_ih = (const float*)d_in[6];
    const float* W_hh = (const float*)d_in[7];
    const float* b_ih = (const float*)d_in[8];
    const float* b_hh = (const float*)d_in[9];

    const int T = 8, F = 128, H = 128;
    const int N = in_sizes[0] / (T * F);
    const int E = in_sizes[1] / (2 * T);

    // workspace carve-up (256B aligned)
    char* base = (char*)d_ws;
    size_t off = 0;
    auto carve = [&](size_t bytes) -> char* {
        char* p = base + off;
        off += (bytes + 255) & ~(size_t)255;
        return p;
    };
    unsigned* packW1  = (unsigned*)carve(4 * 8  * 32 * 8 * sizeof(unsigned));
    unsigned* packW2  = (unsigned*)carve(4 * 8  * 32 * 8 * sizeof(unsigned));
    unsigned* packWih = (unsigned*)carve(4 * 32 * 32 * 8 * sizeof(unsigned));
    unsigned* packWhh = (unsigned*)carve(4 * 32 * 32 * 8 * sizeof(unsigned));
    float* dinv = (float*)carve((size_t)N * sizeof(float));
    float* bufA = (float*)carve((size_t)N * H * sizeof(float));          // xW (f32)
    float* agg  = (float*)carve((size_t)N * H * sizeof(float));
    float* hbuf = (float*)carve((size_t)N * H * sizeof(float));
    float* cbuf = (float*)carve((size_t)N * H * sizeof(float));
    unsigned short* xbf  = (unsigned short*)carve((size_t)N * H * 2);    // bf16 input
    unsigned short* bufB = (unsigned short*)carve((size_t)N * H * 2);    // bf16 relu out
    unsigned short* hbf  = (unsigned short*)carve((size_t)N * H * 2);    // bf16 hidden
    (void)ws_size; (void)n_in; (void)out_size;

    const long long NH = (long long)N * H;
    const int gNHp  = (int)((NH / 2 + 255) / 256);     // pair-granularity grids
    const int gGemm = ((N + 15) / 16 + 7) / 8;         // 8 strips/block
    const int gEdge = (E + 7) / 8;                     // wave per edge
    const int gE256 = (E + 255) / 256;
    const int gN256 = (N + 255) / 256;
    const int gLstm = (N + 31) / 32;

    // One-time (per launch) weight packing + state init
    hipLaunchKernelGGL(pack_weights_kernel, dim3(32),  dim3(256), 0, stream, W1,   packW1,  128, 0, 128);
    hipLaunchKernelGGL(pack_weights_kernel, dim3(32),  dim3(256), 0, stream, W2,   packW2,  128, 0, 128);
    hipLaunchKernelGGL(pack_weights_kernel, dim3(128), dim3(256), 0, stream, W_ih, packWih, 512, 1, 128);
    hipLaunchKernelGGL(pack_weights_kernel, dim3(128), dim3(256), 0, stream, W_hh, packWhh, 512, 1, 128);
    hipLaunchKernelGGL(fill_f32_kernel, dim3(2048), dim3(256), 0, stream, hbuf, 0.0f, NH);
    hipLaunchKernelGGL(fill_f32_kernel, dim3(2048), dim3(256), 0, stream, cbuf, 0.0f, NH);
    hipLaunchKernelGGL(fill_f32_kernel, dim3(2048), dim3(256), 0, stream, (float*)hbf, 0.0f, NH / 2);

    for (int t = 0; t < T; ++t) {
        const float* x_t  = node_features + (size_t)t * N * F;
        const int* src_t  = edge_index + (size_t)t * 2 * E;
        const int* dst_t  = src_t + E;

        // per-timestep symmetric normalization
        hipLaunchKernelGGL(fill_f32_kernel, dim3(256), dim3(256), 0, stream, dinv, 1.0f, (long long)N);
        hipLaunchKernelGGL(deg_count_kernel, dim3(gE256), dim3(256), 0, stream, dinv, dst_t, E);
        hipLaunchKernelGGL(finish_dinv_kernel, dim3(gN256), dim3(256), 0, stream, dinv, N);

        // ---- GCN layer 1 ----
        hipLaunchKernelGGL(cvt_f32_bf16_kernel, dim3(2048), dim3(256), 0, stream, x_t, (unsigned*)xbf, NH / 2);
        hipLaunchKernelGGL(gemm_n128_kernel, dim3(gGemm), dim3(256), 0, stream, xbf, packW1, bufA, N);
        hipLaunchKernelGGL(fill_f32_kernel, dim3(2048), dim3(256), 0, stream, agg, 0.0f, NH);
        hipLaunchKernelGGL(edge_scatter_kernel, dim3(gEdge), dim3(256), 0, stream, bufA, agg, src_t, dst_t, dinv, E);
        hipLaunchKernelGGL(gcn_combine_kernel, dim3(gNHp), dim3(256), 0, stream, agg, bufA, dinv, b1, (unsigned*)bufB, N);

        // ---- GCN layer 2 ----
        hipLaunchKernelGGL(gemm_n128_kernel, dim3(gGemm), dim3(256), 0, stream, bufB, packW2, bufA, N);
        hipLaunchKernelGGL(fill_f32_kernel, dim3(2048), dim3(256), 0, stream, agg, 0.0f, NH);
        hipLaunchKernelGGL(edge_scatter_kernel, dim3(gEdge), dim3(256), 0, stream, bufA, agg, src_t, dst_t, dinv, E);
        hipLaunchKernelGGL(gcn_combine_kernel, dim3(gNHp), dim3(256), 0, stream, agg, bufA, dinv, b2, (unsigned*)bufB, N);

        // ---- LSTM step (fused dual-GEMM + cell) ----
        hipLaunchKernelGGL(lstm_step_kernel, dim3(gLstm), dim3(256), 0, stream,
                           bufB, hbf, packWih, packWhh, b_ih, b_hh, hbuf, cbuf, hbf, N);
    }

    hipLaunchKernelGGL(copy_f32_kernel, dim3(2048), dim3(256), 0, stream,
                       hbuf, (float*)d_out, NH);
}
